// RPN_38671885533230
// MI455X (gfx1250) — matrix operand order repaired and checked by reading, not compiled, not run on hardware
//
#include <hip/hip_runtime.h>
#include <hip/hip_bf16.h>

typedef _Float16 half8  __attribute__((ext_vector_type(8)));
typedef _Float16 half16 __attribute__((ext_vector_type(16)));
typedef float    float8 __attribute__((ext_vector_type(8)));
typedef int      v4i    __attribute__((ext_vector_type(4)));

#define SCALE_CLAMP 4.135166556742356f  // ln(1000/16)

// ---- CDNA5 async global->LDS path (guarded; falls back to sync copies) ----
#if defined(__gfx1250__) && \
    __has_builtin(__builtin_amdgcn_global_load_async_to_lds_b128) && \
    __has_builtin(__builtin_amdgcn_s_wait_asynccnt)
#define USE_ASYNC_LDS 1
typedef __attribute__((address_space(1))) v4i as1_v4i;
typedef __attribute__((address_space(3))) v4i as3_v4i;
#define ASYNC_COPY_B128(gp, lp)                                              \
    __builtin_amdgcn_global_load_async_to_lds_b128(                          \
        (as1_v4i*)(unsigned long long)(const void*)(gp), (as3_v4i*)(lp), 0, 0)
#define ASYNC_WAIT(n) __builtin_amdgcn_s_wait_asynccnt(n)
#else
#define USE_ASYNC_LDS 0
#define ASYNC_COPY_B128(gp, lp) (*(half8*)(lp) = *(const half8*)(gp))
#define ASYNC_WAIT(n) ((void)0)
#endif

__device__ __forceinline__ unsigned mapf(float s) {
    unsigned u = __float_as_uint(s);
    return (u & 0x80000000u) ? ~u : (u | 0x80000000u);
}
__device__ __forceinline__ float unmapf(unsigned k) {
    return __uint_as_float((k & 0x80000000u) ? (k ^ 0x80000000u) : ~k);
}

// ---------------------------------------------------------------------------
// Kernel 0: weight prep.
//   Bw[co][k] f16, k = (ky*3+kx)*256 + ci    (B-matrix rows = N, contiguous K)
//   Whead[n][ci] f16, n: 0..2 = obj, 3..14 = delta(a*4+j), 15 = zero pad
// ---------------------------------------------------------------------------
__global__ void rpn_prep_kernel(const float* __restrict__ conv_w,
                                const float* __restrict__ obj_w,
                                const float* __restrict__ obj_b,
                                const float* __restrict__ delta_w,
                                const float* __restrict__ delta_b,
                                _Float16* __restrict__ Bw,
                                _Float16* __restrict__ Whead,
                                float* __restrict__ biasHead) {
    int i = blockIdx.x * 256 + threadIdx.x;
    if (i < 2304 * 256) {
        int co = i / 2304, k = i % 2304;
        int r = k >> 8, ci = k & 255;
        int ky = r / 3, kx = r % 3;
        Bw[i] = (_Float16)conv_w[((co * 256 + ci) * 3 + ky) * 3 + kx];
    }
    if (i < 16 * 256) {
        int n = i >> 8, ci = i & 255;
        float v = 0.f;
        if (n < 3) v = obj_w[n * 256 + ci];
        else if (n < 15) v = delta_w[(n - 3) * 256 + ci];
        Whead[i] = (_Float16)v;
    }
    if (i < 16) {
        float v = 0.f;
        if (i < 3) v = obj_b[i];
        else if (i < 15) v = delta_b[i - 3];
        biasHead[i] = v;
    }
}

// ---------------------------------------------------------------------------
// Kernel 1: fused 3x3 conv (implicit GEMM via WMMA) + ReLU + 1x1 heads +
//           anchor/delta decode. One WG = 64 pixels x all 256 channels.
//   8 waves: wave w owns N-channels [w*32, w*32+32) as two 16-wide tiles,
//   and all four 16-pixel M tiles. K = 72 steps of 32 (9 taps x 8 ci-chunks).
//   B weight slabs (16KB/step) are staged to LDS with double-buffered async
//   copies (GLOBAL_LOAD_ASYNC_TO_LDS_B128 + s_wait_asynccnt) on CDNA5.
// ---------------------------------------------------------------------------
__global__ void __launch_bounds__(256)
rpn_conv_head_kernel(const float* __restrict__ feat,
                     const float* __restrict__ conv_b,
                     const _Float16* __restrict__ Bw,
                     const _Float16* __restrict__ Whead,
                     const float* __restrict__ biasHead,
                     float* __restrict__ scoresOut,
                     float* __restrict__ boxesOut,
                     int wShift, float stride, float asize) {
    const int W = 1 << wShift, H = W, HW = H * W;
    const int img = blockIdx.y;
    const int tileBase = blockIdx.x * 64;

    __shared__ __align__(16) _Float16 ldsA[64 * 32];   // im2col A tile (4KB)
    // 32KB pool: main loop = double-buffered B slab; epilogue = ReLU tile.
    __shared__ __align__(16) char ldsPool[32768];
    __shared__ __align__(16) float ldsHead[64 * 16];   // head outputs (4KB)
    _Float16* ldsB    = (_Float16*)ldsPool;            // [2][256*32]
    _Float16* ldsRelu = (_Float16*)ldsPool;            // [64*256] (after loop)

    const int lane = threadIdx.x & 31;
    const int wv   = threadIdx.x >> 5;
    const int mrow = lane & 15;
    const int kbA  = (lane < 16) ? 0 : 8;   // 16-bit A-matrix per-lane K base
    const int kbB  = (lane < 16) ? 0 : 8;   // 16-bit B-matrix per-lane K base

    // Stage the 256x32 f16 B slab for k-step s into ldsB buffer `buf`.
    // 1024 x 16B chunks, 4 per thread.
    auto issueB = [&](int s, int buf) {
        #pragma unroll
        for (int e = 0; e < 4; ++e) {
            int c = (e << 8) + (int)threadIdx.x;  // 0..1023
            int row = c >> 2, sub = c & 3;
            const _Float16* gp = Bw + (size_t)row * 2304 + s * 32 + sub * 8;
            _Float16* lp = ldsB + buf * 8192 + row * 32 + sub * 8;
            ASYNC_COPY_B128(gp, lp);
        }
    };

    float8 acc[4][2] = {};
    issueB(0, 0);

    for (int s = 0; s < 72; ++s) {
        const int r = s >> 3, ci0 = (s & 7) << 5;
        const int ky = r / 3 - 1, kx = r % 3 - 1;
        __syncthreads();  // prior readers of ldsA and ldsB[(s+1)&1] are done
        if (s + 1 < 72) issueB(s + 1, (s + 1) & 1);

        // Stage A tile: im2col gather, f32 -> f16.
        const float* fbase = feat + ((size_t)img * 256 + ci0) * HW;
        #pragma unroll
        for (int e = 0; e < 8; ++e) {
            int idx = threadIdx.x + (e << 8);  // 0..2047 over 64x32 tile
            int pix = idx & 63, c = idx >> 6;
            int p = tileBase + pix;
            int y = (p >> wShift) + ky;
            int x = (p & (W - 1)) + kx;
            float v = 0.f;
            if ((unsigned)y < (unsigned)H && (unsigned)x < (unsigned)W)
                v = fbase[(size_t)c * HW + (y << wShift) + x];
            ldsA[pix * 32 + c] = (_Float16)v;
        }
        if (s + 1 < 72) {
            int ci0n = ((s + 1) & 7) << 5;
            __builtin_prefetch(feat + ((size_t)img * 256 + ci0n) * HW + tileBase, 0, 1);
            ASYNC_WAIT(4);   // batch s retired (in-order); batch s+1 in flight
        } else {
            ASYNC_WAIT(0);   // last step: drain everything
        }
        __syncthreads();

        const _Float16* ldsBs = ldsB + (s & 1) * 8192;
        half16 bf[2];
        #pragma unroll
        for (int nt = 0; nt < 2; ++nt) {
            const _Float16* bp = &ldsBs[(wv * 32 + nt * 16 + mrow) * 32 + kbB];
            half8 lo = *(const half8*)bp;
            half8 hi = *(const half8*)(bp + 16);
            #pragma unroll
            for (int i = 0; i < 8; ++i) { bf[nt][i] = lo[i]; bf[nt][8 + i] = hi[i]; }
        }
        #pragma unroll
        for (int mt = 0; mt < 4; ++mt) {
            const _Float16* ap = &ldsA[(mt * 16 + mrow) * 32 + kbA];
            half8 lo = *(const half8*)ap;
            half8 hi = *(const half8*)(ap + 16);
            half16 af;
            #pragma unroll
            for (int i = 0; i < 8; ++i) { af[i] = lo[i]; af[8 + i] = hi[i]; }
            #pragma unroll
            for (int nt = 0; nt < 2; ++nt)
                acc[mt][nt] = __builtin_amdgcn_wmma_f32_16x16x32_f16(
                    false, af, false, bf[nt], (short)0, acc[mt][nt], false, false);
        }
    }

    // Epilogue 1: bias + ReLU -> LDS (f16). C/D layout: vgpr rr, lane l ->
    //   M = rr + 8*(l>=16), N = l%16.
    __syncthreads();  // ldsB lifetime over; pool becomes ldsRelu
    const int mhi = (lane >> 4) << 3;
    #pragma unroll
    for (int mt = 0; mt < 4; ++mt)
        #pragma unroll
        for (int nt = 0; nt < 2; ++nt) {
            int n = wv * 32 + nt * 16 + mrow;
            float cb = conv_b[n];
            #pragma unroll
            for (int rr = 0; rr < 8; ++rr) {
                int m = mt * 16 + mhi + rr;
                float v = acc[mt][nt][rr] + cb;
                ldsRelu[m * 256 + n] = (_Float16)fmaxf(v, 0.f);
            }
        }
    __syncthreads();

    // Epilogue 2: head GEMM (64 px x 16 out x K=256) on waves 0..3.
    if (wv < 4) {
        float8 hacc = {};
        #pragma unroll
        for (int ks = 0; ks < 8; ++ks) {
            const _Float16* ap = &ldsRelu[(wv * 16 + mrow) * 256 + ks * 32 + kbA];
            half8 alo = *(const half8*)ap;
            half8 ahi = *(const half8*)(ap + 16);
            const _Float16* bp = Whead + mrow * 256 + ks * 32 + kbB;
            half8 blo = *(const half8*)bp;
            half8 bhi = *(const half8*)(bp + 16);
            half16 af, bfh;
            #pragma unroll
            for (int i = 0; i < 8; ++i) {
                af[i] = alo[i]; af[8 + i] = ahi[i];
                bfh[i] = blo[i]; bfh[8 + i] = bhi[i];
            }
            hacc = __builtin_amdgcn_wmma_f32_16x16x32_f16(
                false, af, false, bfh, (short)0, hacc, false, false);
        }
        float hb = biasHead[mrow];
        #pragma unroll
        for (int rr = 0; rr < 8; ++rr) {
            int m = wv * 16 + mhi + rr;
            ldsHead[m * 16 + mrow] = hacc[rr] + hb;
        }
    }
    __syncthreads();

    // Epilogue 3: anchors + Box2Box decode, write scores/boxes.
    if (threadIdx.x < 192) {
        int pl = threadIdx.x / 3, a = threadIdx.x % 3;
        int p = tileBase + pl;
        float score = ldsHead[pl * 16 + a];
        float d0 = ldsHead[pl * 16 + 3 + a * 4 + 0];
        float d1 = ldsHead[pl * 16 + 3 + a * 4 + 1];
        float d2 = ldsHead[pl * 16 + 3 + a * 4 + 2];
        float d3 = ldsHead[pl * 16 + 3 + a * 4 + 3];
        float ratio = (a == 0) ? 0.5f : ((a == 1) ? 1.0f : 2.0f);
        float wa = sqrtf(asize * asize / ratio);
        float ha = wa * ratio;
        float sx = (float)(p & (W - 1)) * stride;
        float sy = (float)(p >> wShift) * stride;
        float dw = fminf(d2, SCALE_CLAMP), dh = fminf(d3, SCALE_CLAMP);
        float pcx = d0 * wa + sx, pcy = d1 * ha + sy;
        float pw = __expf(dw) * wa, ph = __expf(dh) * ha;
        size_t oi = (size_t)img * (size_t)(HW * 3) + (size_t)(p * 3 + a);
        scoresOut[oi] = score;
        boxesOut[oi * 4 + 0] = pcx - 0.5f * pw;
        boxesOut[oi * 4 + 1] = pcy - 0.5f * ph;
        boxesOut[oi * 4 + 2] = pcx + 0.5f * pw;
        boxesOut[oi * 4 + 3] = pcy + 0.5f * ph;
    }
}

// ---------------------------------------------------------------------------
// Kernel 2: per-(image,level) top-1000 via 2-pass radix select on sortable keys.
// ---------------------------------------------------------------------------
__global__ void __launch_bounds__(512)
rpn_topk_kernel(const float* __restrict__ scoresL,
                const float* __restrict__ boxesL,
                float* __restrict__ combScores,
                float* __restrict__ combBoxes,
                int K, int lvl) {
    const int img = blockIdx.x;
    const int tid = threadIdx.x;
    const float* sc = scoresL + (size_t)img * K;
    const float* bx = boxesL + (size_t)img * K * 4;
    __shared__ int hist[256];
    __shared__ int binA, aboveA, thr16, cA, cB;

    for (int b = tid; b < 256; b += 512) hist[b] = 0;
    __syncthreads();
    for (int i = tid; i < K; i += 512) atomicAdd(&hist[mapf(sc[i]) >> 24], 1);
    __syncthreads();
    if (tid == 0) {
        int cum = 0, b = 255;
        for (; b > 0; --b) { if (cum + hist[b] >= 1000) break; cum += hist[b]; }
        binA = b; aboveA = cum;
    }
    __syncthreads();
    int bA = binA;
    for (int b = tid; b < 256; b += 512) hist[b] = 0;
    __syncthreads();
    for (int i = tid; i < K; i += 512) {
        unsigned k = mapf(sc[i]);
        if ((int)(k >> 24) == bA) atomicAdd(&hist[(k >> 16) & 255], 1);
    }
    __syncthreads();
    if (tid == 0) {
        int cum = aboveA, b = 255;
        for (; b > 0; --b) { if (cum + hist[b] >= 1000) break; cum += hist[b]; }
        thr16 = (bA << 8) | b;
        cA = 0; cB = cum;  // definite slots [0,cum), ties fill [cum,1000)
    }
    __syncthreads();
    int t16 = thr16;
    float* cs = combScores + (size_t)img * 3000 + lvl * 1000;
    float* cb = combBoxes + ((size_t)img * 3000 + (size_t)lvl * 1000) * 4;
    for (int i = tid; i < K; i += 512) {
        unsigned k = mapf(sc[i]);
        int kh = (int)(k >> 16);
        int slot = -1;
        if (kh > t16) slot = atomicAdd(&cA, 1);
        else if (kh == t16) { int t = atomicAdd(&cB, 1); if (t < 1000) slot = t; }
        if (slot >= 0 && slot < 1000) {
            cs[slot] = sc[i];
            cb[slot * 4 + 0] = bx[i * 4 + 0];
            cb[slot * 4 + 1] = bx[i * 4 + 1];
            cb[slot * 4 + 2] = bx[i * 4 + 2];
            cb[slot * 4 + 3] = bx[i * 4 + 3];
        }
    }
}

// ---------------------------------------------------------------------------
// Kernel 3: per-image NMS. Clip+validate, bitonic sort 3000 (pad 4096) keys
// descending in LDS, greedy suppression (cross-level IOU == 0 by lvl offset
// semantics), emit first 1000 kept as (x1,y1,x2,y2,score), zeros after.
// ---------------------------------------------------------------------------
__device__ __forceinline__ float clipc(float v) {
    return fminf(fmaxf(v, 0.f), 1024.f);
}

__global__ void __launch_bounds__(1024)
rpn_nms_kernel(const float* __restrict__ combScores,
               const float* __restrict__ combBoxes,
               float* __restrict__ out) {
    const int img = blockIdx.x;
    const int tid = threadIdx.x;
    const float* cs = combScores + (size_t)img * 3000;
    const float* cb = combBoxes + (size_t)img * 3000 * 4;

    __shared__ unsigned keys[4096];
    __shared__ unsigned short vidx[4096];
    __shared__ unsigned char sup[3008];
    __shared__ unsigned short outIdx[1000];
    __shared__ int nKept;

    for (int i = tid; i < 4096; i += 1024) {
        unsigned key = 0u;
        unsigned short id = 0xFFFFu;
        if (i < 3000) {
            float x1 = clipc(cb[i * 4 + 0]);
            float y1 = clipc(cb[i * 4 + 1]);
            float x2 = clipc(cb[i * 4 + 2]);
            float y2 = clipc(cb[i * 4 + 3]);
            bool valid = (x2 - x1 > 0.f) && (y2 - y1 > 0.f);
            float s = valid ? cs[i] : -INFINITY;
            key = mapf(s);
            id = (unsigned short)i;
        }
        keys[i] = key; vidx[i] = id;
    }
    __syncthreads();

    for (int k = 2; k <= 4096; k <<= 1) {
        for (int j = k >> 1; j > 0; j >>= 1) {
            for (int i = tid; i < 4096; i += 1024) {
                int ixj = i ^ j;
                if (ixj > i) {
                    unsigned a = keys[i], b = keys[ixj];
                    bool up = ((i & k) == 0);
                    bool sw = up ? (a < b) : (a > b);  // overall descending
                    if (sw) {
                        keys[i] = b; keys[ixj] = a;
                        unsigned short t = vidx[i]; vidx[i] = vidx[ixj]; vidx[ixj] = t;
                    }
                }
            }
            __syncthreads();
        }
    }

    const unsigned NEGINF_KEY = 0x007FFFFFu;  // mapf(-inf)
    for (int i = tid; i < 3008; i += 1024)
        sup[i] = (i < 3000) ? ((keys[i] == NEGINF_KEY) ? 1 : 0) : 1;
    __syncthreads();

    for (int i = 0; i < 3000; ++i) {
        __syncthreads();
        if (sup[i]) continue;  // uniform (LDS) -> barrier-safe
        int idi = vidx[i];
        float ax1 = clipc(cb[idi * 4 + 0]);
        float ay1 = clipc(cb[idi * 4 + 1]);
        float ax2 = clipc(cb[idi * 4 + 2]);
        float ay2 = clipc(cb[idi * 4 + 3]);
        float aarea = (ax2 - ax1) * (ay2 - ay1);
        int li = idi / 1000;
        for (int j = i + 1 + tid; j < 3000; j += 1024) {
            if (sup[j]) continue;
            int idj = vidx[j];
            if (idj / 1000 != li) continue;  // level offset => IOU == 0
            float bx1 = clipc(cb[idj * 4 + 0]);
            float by1 = clipc(cb[idj * 4 + 1]);
            float bx2 = clipc(cb[idj * 4 + 2]);
            float by2 = clipc(cb[idj * 4 + 3]);
            float iw = fmaxf(fminf(ax2, bx2) - fmaxf(ax1, bx1), 0.f);
            float ih = fmaxf(fminf(ay2, by2) - fmaxf(ay1, by1), 0.f);
            float inter = iw * ih;
            float barea = (bx2 - bx1) * (by2 - by1);
            float iou = inter / fmaxf(aarea + barea - inter, 1e-9f);
            if (iou > 0.7f) sup[j] = 1;
        }
    }
    __syncthreads();

    if (tid == 0) {
        int r = 0;
        for (int i = 0; i < 3000 && r < 1000; ++i)
            if (!sup[i]) outIdx[r++] = (unsigned short)i;
        nKept = r;
    }
    __syncthreads();

    float* op = out + (size_t)img * 1000 * 5;
    for (int r = tid; r < 1000; r += 1024) {
        float o0 = 0.f, o1 = 0.f, o2 = 0.f, o3 = 0.f, o4 = 0.f;
        if (r < nKept) {
            int i = outIdx[r];
            int id = vidx[i];
            o0 = clipc(cb[id * 4 + 0]);
            o1 = clipc(cb[id * 4 + 1]);
            o2 = clipc(cb[id * 4 + 2]);
            o3 = clipc(cb[id * 4 + 3]);
            o4 = unmapf(keys[i]);
        }
        op[r * 5 + 0] = o0; op[r * 5 + 1] = o1; op[r * 5 + 2] = o2;
        op[r * 5 + 3] = o3; op[r * 5 + 4] = o4;
    }
}

// ---------------------------------------------------------------------------
extern "C" void kernel_launch(void* const* d_in, const int* in_sizes, int n_in,
                              void* d_out, int out_size, void* d_ws, size_t ws_size,
                              hipStream_t stream) {
    const float* feat[3] = {(const float*)d_in[0], (const float*)d_in[1], (const float*)d_in[2]};
    const float* conv_w  = (const float*)d_in[3];
    const float* conv_b  = (const float*)d_in[4];
    const float* obj_w   = (const float*)d_in[5];
    const float* obj_b   = (const float*)d_in[6];
    const float* delta_w = (const float*)d_in[7];
    const float* delta_b = (const float*)d_in[8];
    (void)in_sizes; (void)n_in; (void)out_size; (void)ws_size;

    char* ws = (char*)d_ws;
    size_t off = 0;
    auto take = [&](size_t bytes) {
        char* p = ws + off;
        off = (off + bytes + 255) & ~(size_t)255;
        return p;
    };
    _Float16* Bw       = (_Float16*)take((size_t)2304 * 256 * 2);
    _Float16* Whead    = (_Float16*)take((size_t)16 * 256 * 2);
    float*    biasHead = (float*)take(16 * 4);

    const int HWs[3] = {128 * 128, 64 * 64, 32 * 32};
    float* scoresL[3]; float* boxesL[3];
    for (int l = 0; l < 3; ++l) {
        size_t K = (size_t)HWs[l] * 3;
        scoresL[l] = (float*)take(4 * K * 4);
        boxesL[l]  = (float*)take(4 * K * 16);
    }
    float* combScores = (float*)take((size_t)4 * 3000 * 4);
    float* combBoxes  = (float*)take((size_t)4 * 3000 * 16);

    rpn_prep_kernel<<<(2304 * 256 + 255) / 256, 256, 0, stream>>>(
        conv_w, obj_w, obj_b, delta_w, delta_b, Bw, Whead, biasHead);

    const int   wShifts[3] = {7, 6, 5};
    const float strides[3] = {8.f, 16.f, 32.f};
    const float sizes[3]   = {32.f, 64.f, 128.f};
    for (int l = 0; l < 3; ++l) {
        dim3 grid(HWs[l] / 64, 4);
        rpn_conv_head_kernel<<<grid, 256, 0, stream>>>(
            feat[l], conv_b, Bw, Whead, biasHead,
            scoresL[l], boxesL[l], wShifts[l], strides[l], sizes[l]);
    }
    for (int l = 0; l < 3; ++l)
        rpn_topk_kernel<<<4, 512, 0, stream>>>(
            scoresL[l], boxesL[l], combScores, combBoxes, HWs[l] * 3, l);

    rpn_nms_kernel<<<4, 1024, 0, stream>>>(combScores, combBoxes, (float*)d_out);
}
